// IBSNet_transformer_66571993088198
// MI455X (gfx1250) — matrix operand order repaired and checked by reading, not compiled, not run on hardware
//
#include <hip/hip_runtime.h>

typedef __bf16 bf16;
typedef __attribute__((ext_vector_type(16))) __bf16 v16bf;
typedef __attribute__((ext_vector_type(8)))  __bf16 v8bf;
typedef __attribute__((ext_vector_type(8)))  float  v8f;

#define NPTS 4096
#define KNN  16
#define DIM  128
#define HID  512
#define PH   64

// ---------------------------------------------------------------- WMMA helpers

__device__ __forceinline__ v8f wmma_bf16(v16bf a, v16bf b, v8f c) {
    // D = A(16x32 bf16) * B(32x16 bf16) + C(16x16 f32)
    return __builtin_amdgcn_wmma_f32_16x16x32_bf16(
        false, a, false, b, (short)0, c, false, false);
}

// A-fragment for 16x16x32 bf16 WMMA from row-major bf16 matrix W[*, ldk].
// lane = m + 16*khalf holds 16 contiguous K values khalf*16..khalf*16+15.
__device__ __forceinline__ v16bf load_afrag(const bf16* __restrict__ W,
                                            int row, int ldk, int k0, int khalf) {
    const bf16* p = W + (size_t)row * ldk + k0 + khalf * 16;
    v8bf lo = *(const v8bf*)p;
    v8bf hi = *(const v8bf*)(p + 8);
    v16bf a;
#pragma unroll
    for (int i = 0; i < 8; ++i) { a[i] = lo[i]; a[i + 8] = hi[i]; }
    return a;
}

// B-fragment: lane = n + 16*khalf holds 16 contiguous K values at khalf*16.
// Staged LDS layout is [col][K], so this is a single contiguous 32B load.
__device__ __forceinline__ v16bf load_bfrag(const bf16* p) {
    return *(const v16bf*)p;
}

// ---------------------------------------------------------------- prep kernel
// Convert fused-path weights to bf16; fold conv-bias + BN into scale/shift.

__global__ __launch_bounds__(256) void prep_kernel(
    const float* __restrict__ w_a1, const float* __restrict__ w_a2,
    const float* __restrict__ w_e,  const float* __restrict__ w_p2,
    const float* __restrict__ b_p1, const float* __restrict__ g1,
    const float* __restrict__ be1,  const float* __restrict__ m1,
    const float* __restrict__ v1,
    const float* __restrict__ b_a1, const float* __restrict__ g2,
    const float* __restrict__ be2,  const float* __restrict__ m2,
    const float* __restrict__ v2,
    bf16* __restrict__ w_a1b, bf16* __restrict__ w_a2b,
    bf16* __restrict__ w_eb,  bf16* __restrict__ w_p2b,
    float* __restrict__ scale1, float* __restrict__ bias1e,
    float* __restrict__ scale2, float* __restrict__ bias2e) {
    int i = blockIdx.x * 256 + threadIdx.x;
    if (i < HID * DIM) {
        w_a1b[i] = (bf16)w_a1[i];
        w_a2b[i] = (bf16)w_a2[i];
        w_eb[i]  = (bf16)w_e[i];
    }
    if (i < DIM * PH) w_p2b[i] = (bf16)w_p2[i];
    if (i < PH) {
        float s = g1[i] * rsqrtf(v1[i] + 1e-5f);
        scale1[i]  = s;
        bias1e[i]  = s * (b_p1[i] - m1[i]) + be1[i];
    }
    if (i < HID) {
        float s = g2[i] * rsqrtf(v2[i] + 1e-5f);
        scale2[i]  = s;
        bias2e[i]  = s * (b_a1[i] - m2[i]) + be2[i];
    }
}

// ---------------------------------------------------------------- KNN kernel
// One wave per query point; positions of the batch staged in LDS (48 KB).

__global__ __launch_bounds__(256) void knn_kernel(const float* __restrict__ pos,
                                                  int* __restrict__ idx_out) {
    __shared__ float sx[NPTS], sy[NPTS], sz[NPTS];
    int b     = blockIdx.x / (NPTS / 8);
    int pbase = (blockIdx.x % (NPTS / 8)) * 8;
    const float* pb = pos + (size_t)b * 3 * NPTS;
    for (int i = threadIdx.x; i < NPTS; i += 256) {
        sx[i] = pb[i];
        sy[i] = pb[NPTS + i];
        sz[i] = pb[2 * NPTS + i];
    }
    __syncthreads();
    int wave = threadIdx.x >> 5, lane = threadIdx.x & 31;
    int n = pbase + wave;
    float cx = sx[n], cy = sy[n], cz = sz[n];
    float cd[KNN];
    int   ci[KNN];
#pragma unroll
    for (int i = 0; i < KNN; ++i) { cd[i] = 1e30f; ci[i] = 0; }
    float wmax = 1e30f;
    int   wslot = 0;
    for (int m = lane; m < NPTS; m += 32) {
        float dx = cx - sx[m], dy = cy - sy[m], dz = cz - sz[m];
        float d  = dx * dx + dy * dy + dz * dz;
        if (d < wmax) {
            cd[wslot] = d;
            ci[wslot] = m;
            wmax = -1e30f;
#pragma unroll
            for (int i = 0; i < KNN; ++i)
                if (cd[i] > wmax) { wmax = cd[i]; wslot = i; }
        }
    }
    int* out = idx_out + ((size_t)b * NPTS + n) * KNN;
    for (int r = 0; r < KNN; ++r) {
        float bv = 1e30f;
        int   bs = 0;
#pragma unroll
        for (int i = 0; i < KNN; ++i)
            if (cd[i] < bv) { bv = cd[i]; bs = i; }
        float rv   = bv;
        int   ridx = ci[bs];
        int   rcode = lane * KNN + bs;
#pragma unroll
        for (int s = 16; s >= 1; s >>= 1) {
            float ov = __shfl_xor(rv, s);
            int   oi = __shfl_xor(ridx, s);
            int   oc = __shfl_xor(rcode, s);
            if (ov < rv || (ov == rv && oc < rcode)) { rv = ov; ridx = oi; rcode = oc; }
        }
        if (lane == rcode / KNN) cd[rcode % KNN] = 1e30f;
        if (lane == 0) out[r] = ridx;
    }
}

// ---------------------------------------------------------------- GEMM (M=128)
// Y[128, 16384] = W[128,K] @ X[B,K,4096] + bias.  Block tile 128x32, 8 waves,
// each wave one 16-row slab x two 16-col WMMA tiles. Optional f32 / transposed
// bf16 / transposed f32 outputs.

__global__ __launch_bounds__(256) void gemm128_kernel(
    const float* __restrict__ W, const float* __restrict__ X,
    const float* __restrict__ bias, int Kdim,
    float* __restrict__ Yf, bf16* __restrict__ YtBf, float* __restrict__ YtF) {
    __shared__ bf16 wt[128 * 32];   // [row][k]
    __shared__ bf16 xt[32 * 32];    // [col][k]
    int colg = blockIdx.x * 32;
    int b    = colg >> 12;
    int n0   = colg & (NPTS - 1);
    const float* Xb = X + (size_t)b * Kdim * NPTS;
    int tid = threadIdx.x, wave = tid >> 5, lane = tid & 31;
    int m = lane & 15, khalf = lane >> 4;
    v8f accs[2] = {{}, {}};
    for (int k0 = 0; k0 < Kdim; k0 += 32) {
        {   // stage W tile, f32 -> bf16
            int row = tid >> 1;
            int kk  = (tid & 1) * 16;
            const float* src = W + (size_t)row * Kdim + k0 + kk;
#pragma unroll
            for (int i = 0; i < 16; ++i) wt[row * 32 + kk + i] = (bf16)src[i];
        }
        {   // stage X tile transposed to [col][k]
            int col = tid >> 3;
            int kk  = (tid & 7) * 4;
            const float* src = Xb + (size_t)(k0 + kk) * NPTS + n0 + col;
#pragma unroll
            for (int i = 0; i < 4; ++i)
                xt[col * 32 + kk + i] = (bf16)src[(size_t)i * NPTS];
        }
        __syncthreads();
        v16bf a = load_afrag(wt, 16 * wave + m, 32, 0, khalf);
        v16bf b0 = load_bfrag(&xt[m * 32 + khalf * 16]);
        v16bf b1 = load_bfrag(&xt[(16 + m) * 32 + khalf * 16]);
        accs[0] = wmma_bf16(a, b0, accs[0]);
        accs[1] = wmma_bf16(a, b1, accs[1]);
        __syncthreads();
    }
    int chb = 16 * wave + 8 * khalf;
#pragma unroll
    for (int t = 0; t < 2; ++t) {
        int col = n0 + t * 16 + m;
        size_t colg2 = (size_t)b * NPTS + col;
        if (Yf) {
#pragma unroll
            for (int r = 0; r < 8; ++r)
                Yf[(size_t)b * DIM * NPTS + (size_t)(chb + r) * NPTS + col] =
                    accs[t][r] + bias[chb + r];
        }
        if (YtBf) {
            v8bf o;
#pragma unroll
            for (int r = 0; r < 8; ++r) o[r] = (bf16)(accs[t][r] + bias[chb + r]);
            *(v8bf*)&YtBf[colg2 * DIM + chb] = o;
        }
        if (YtF) {
#pragma unroll
            for (int r = 0; r < 8; ++r)
                YtF[colg2 * DIM + chb + r] = accs[t][r] + bias[chb + r];
        }
    }
}

// ---------------------------------------------------------------- fused kernel
// Per block: 32 consecutive points. Per iteration: 2 points x 16 neighbors =
// 32 columns. pos-MLP + attn-MLP via WMMA, softmax-over-K via xor shuffles,
// aggregation, then one final w_e GEMM over the 32 accumulated points.

__global__ __launch_bounds__(256) void fused_attn_kernel(
    const float* __restrict__ pos,    // [B,3,4096]
    const int*   __restrict__ knn,    // [B,4096,16]
    const bf16*  __restrict__ keyT,   // [B,4096,128]
    const bf16*  __restrict__ queryT, // [B,4096,128]
    const float* __restrict__ valueT, // [B,4096,128]
    const bf16*  __restrict__ w_p2b, const float* __restrict__ b_p2,
    const float* __restrict__ w_p1,
    const float* __restrict__ scale1, const float* __restrict__ bias1e,
    const bf16*  __restrict__ w_a1b,
    const float* __restrict__ scale2, const float* __restrict__ bias2e,
    const bf16*  __restrict__ w_a2b, const float* __restrict__ b_a2,
    const bf16*  __restrict__ w_eb,  const float* __restrict__ b_e,
    float* __restrict__ out)          // [B,512,4096]
{
    __shared__ bf16  s_xin[32 * DIM];   // qk_rel + pe, [col][k]
    __shared__ bf16  s_pe[32 * DIM];    // pe, [col][ch]
    __shared__ bf16  s_ph1[32 * PH];    // pos-MLP hidden, [col][k]
    __shared__ bf16  s_hid[32 * HID];   // attn-MLP hidden, [col][k]
    __shared__ bf16  s_agg[32 * DIM];   // aggregated features, [pt][ch]
    __shared__ int   s_nbr[32];
    __shared__ float s_prel[32][3];

    int tid = threadIdx.x, wave = tid >> 5, lane = tid & 31;
    int m = lane & 15, khalf = lane >> 4;
    int p0  = blockIdx.x * 32;
    int b   = p0 >> 12;
    int nb0 = p0 & (NPTS - 1);

    for (int it = 0; it < 16; ++it) {
        int cpt[2];
        cpt[0] = nb0 + 2 * it;
        cpt[1] = nb0 + 2 * it + 1;
        if (tid < 32) {
            int pt = tid >> 4;
            int nb = knn[((size_t)b * NPTS + cpt[pt]) * KNN + (tid & 15)];
            s_nbr[tid] = nb;
            const float* pbb = pos + (size_t)b * 3 * NPTS;
            s_prel[tid][0] = pbb[cpt[pt]] - pbb[nb];
            s_prel[tid][1] = pbb[NPTS + cpt[pt]] - pbb[NPTS + nb];
            s_prel[tid][2] = pbb[2 * NPTS + cpt[pt]] - pbb[2 * NPTS + nb];
        }
        __syncthreads();
        {   // pos-MLP layer 1 (3 -> 64) + BN + ReLU, pure VALU (tiny)
            int col   = tid >> 3;
            int hbase = (tid & 7) * 8;
            float px = s_prel[col][0], py = s_prel[col][1], pz = s_prel[col][2];
#pragma unroll
            for (int i = 0; i < 8; ++i) {
                int h = hbase + i;
                float s = w_p1[h * 3] * px + w_p1[h * 3 + 1] * py + w_p1[h * 3 + 2] * pz;
                s = scale1[h] * s + bias1e[h];
                s_ph1[col * PH + h] = (bf16)fmaxf(s, 0.f);
            }
        }
        __syncthreads();
        {   // pe = w_p2 @ ph1 + b_p2  (128x64 GEMM, K=64, WMMA)
            v8f acc[2] = {{}, {}};
#pragma unroll
            for (int ks = 0; ks < 2; ++ks) {
                v16bf a  = load_afrag(w_p2b, 16 * wave + m, PH, ks * 32, khalf);
                v16bf b0 = load_bfrag(&s_ph1[m * PH + ks * 32 + khalf * 16]);
                v16bf b1 = load_bfrag(&s_ph1[(16 + m) * PH + ks * 32 + khalf * 16]);
                acc[0] = wmma_bf16(a, b0, acc[0]);
                acc[1] = wmma_bf16(a, b1, acc[1]);
            }
            int chb = 16 * wave + 8 * khalf;
#pragma unroll
            for (int t = 0; t < 2; ++t) {
                v8bf o;
#pragma unroll
                for (int r = 0; r < 8; ++r) o[r] = (bf16)(acc[t][r] + b_p2[chb + r]);
                *(v8bf*)&s_pe[(t * 16 + m) * DIM + chb] = o;
            }
        }
        __syncthreads();
        {   // xin = query(center) - key(neighbor) + pe   (gather hits L2)
            int col = tid >> 3;
            int cb  = (tid & 7) * 16;
            int pt  = col >> 4;
            int nb  = s_nbr[col];
            const bf16* q  = queryT + ((size_t)b * NPTS + cpt[pt]) * DIM + cb;
            const bf16* k  = keyT   + ((size_t)b * NPTS + nb) * DIM + cb;
            const bf16* pe = &s_pe[col * DIM + cb];
            bf16* x = &s_xin[col * DIM + cb];
#pragma unroll
            for (int i = 0; i < 16; ++i)
                x[i] = (bf16)((float)q[i] - (float)k[i] + (float)pe[i]);
        }
        __syncthreads();
        {   // hidden = relu(bn2(w_a1 @ xin))  (512x128 GEMM, K=128)
            v8f acc[4][2] = {};
#pragma unroll
            for (int ks = 0; ks < 4; ++ks) {
                v16bf bf0 = load_bfrag(&s_xin[m * DIM + ks * 32 + khalf * 16]);
                v16bf bf1 = load_bfrag(&s_xin[(16 + m) * DIM + ks * 32 + khalf * 16]);
#pragma unroll
                for (int rt = 0; rt < 4; ++rt) {
                    v16bf a = load_afrag(w_a1b, 64 * wave + 16 * rt + m, DIM, ks * 32, khalf);
                    acc[rt][0] = wmma_bf16(a, bf0, acc[rt][0]);
                    acc[rt][1] = wmma_bf16(a, bf1, acc[rt][1]);
                }
            }
#pragma unroll
            for (int rt = 0; rt < 4; ++rt) {
                int chb = 64 * wave + 16 * rt + 8 * khalf;
#pragma unroll
                for (int t = 0; t < 2; ++t) {
                    v8bf o;
#pragma unroll
                    for (int r = 0; r < 8; ++r) {
                        float v = scale2[chb + r] * acc[rt][t][r] + bias2e[chb + r];
                        o[r] = (bf16)fmaxf(v, 0.f);
                    }
                    *(v8bf*)&s_hid[(t * 16 + m) * HID + chb] = o;
                }
            }
        }
        __syncthreads();
        {   // logits = w_a2 @ hidden + b_a2 (128x512, K=512); softmax over K; agg
            v8f acc[2] = {{}, {}};
#pragma unroll
            for (int ks = 0; ks < 16; ++ks) {
                v16bf a  = load_afrag(w_a2b, 16 * wave + m, HID, ks * 32, khalf);
                v16bf b0 = load_bfrag(&s_hid[m * HID + ks * 32 + khalf * 16]);
                v16bf b1 = load_bfrag(&s_hid[(16 + m) * HID + ks * 32 + khalf * 16]);
                acc[0] = wmma_bf16(a, b0, acc[0]);
                acc[1] = wmma_bf16(a, b1, acc[1]);
            }
            int chb = 16 * wave + 8 * khalf;
#pragma unroll
            for (int t = 0; t < 2; ++t) {
                const float* vptr  = valueT + ((size_t)b * NPTS + cpt[t]) * DIM + chb;
                const bf16*  peptr = &s_pe[(t * 16 + m) * DIM + chb];
                float aggv[8];
#pragma unroll
                for (int r = 0; r < 8; ++r) {
                    float lg = acc[t][r] + b_a2[chb + r];
                    float mx = lg;
#pragma unroll
                    for (int s = 1; s < 16; s <<= 1) mx = fmaxf(mx, __shfl_xor(mx, s));
                    float e  = __expf(lg - mx);
                    float sm = e;
#pragma unroll
                    for (int s = 1; s < 16; s <<= 1) sm += __shfl_xor(sm, s);
                    float attn = e / sm;
                    float pa = attn * (vptr[r] + (float)peptr[r]);
#pragma unroll
                    for (int s = 1; s < 16; s <<= 1) pa += __shfl_xor(pa, s);
                    aggv[r] = pa;
                }
                if (m == 0) {
                    v8bf o;
#pragma unroll
                    for (int r = 0; r < 8; ++r) o[r] = (bf16)aggv[r];
                    *(v8bf*)&s_agg[(2 * it + t) * DIM + chb] = o;
                }
            }
        }
        __syncthreads();
    }
    {   // out = w_e @ agg + b_e  (512x128 GEMM, K=128, 32 points)
        v8f acc[4][2] = {};
#pragma unroll
        for (int ks = 0; ks < 4; ++ks) {
            v16bf bf0 = load_bfrag(&s_agg[m * DIM + ks * 32 + khalf * 16]);
            v16bf bf1 = load_bfrag(&s_agg[(16 + m) * DIM + ks * 32 + khalf * 16]);
#pragma unroll
            for (int rt = 0; rt < 4; ++rt) {
                v16bf a = load_afrag(w_eb, 64 * wave + 16 * rt + m, DIM, ks * 32, khalf);
                acc[rt][0] = wmma_bf16(a, bf0, acc[rt][0]);
                acc[rt][1] = wmma_bf16(a, bf1, acc[rt][1]);
            }
        }
#pragma unroll
        for (int rt = 0; rt < 4; ++rt) {
            int chb = 64 * wave + 16 * rt + 8 * khalf;
#pragma unroll
            for (int t = 0; t < 2; ++t) {
                int pt = nb0 + t * 16 + m;
#pragma unroll
                for (int r = 0; r < 8; ++r)
                    out[(size_t)b * HID * NPTS + (size_t)(chb + r) * NPTS + pt] =
                        acc[rt][t][r] + b_e[chb + r];
            }
        }
    }
}

// ---------------------------------------------------------------- launch

extern "C" void kernel_launch(void* const* d_in, const int* in_sizes, int n_in,
                              void* d_out, int out_size, void* d_ws, size_t ws_size,
                              hipStream_t stream) {
    (void)in_sizes; (void)n_in; (void)out_size; (void)ws_size;
    const int B = 4;
    const float* obj   = (const float*)d_in[0];
    const float* pos   = (const float*)d_in[1];
    const float* qpts  = (const float*)d_in[2];
    const float* w_sq  = (const float*)d_in[3];  const float* b_sq = (const float*)d_in[4];
    const float* w_so  = (const float*)d_in[5];  const float* b_so = (const float*)d_in[6];
    const float* w_k   = (const float*)d_in[7];  const float* b_k  = (const float*)d_in[8];
    const float* w_q   = (const float*)d_in[9];  const float* b_q  = (const float*)d_in[10];
    const float* w_v   = (const float*)d_in[11]; const float* b_v  = (const float*)d_in[12];
    const float* w_p1  = (const float*)d_in[13]; const float* b_p1 = (const float*)d_in[14];
    const float* g_bn1 = (const float*)d_in[15]; const float* be_bn1 = (const float*)d_in[16];
    const float* m_bn1 = (const float*)d_in[17]; const float* v_bn1  = (const float*)d_in[18];
    const float* w_p2  = (const float*)d_in[19]; const float* b_p2 = (const float*)d_in[20];
    const float* w_a1  = (const float*)d_in[21]; const float* b_a1 = (const float*)d_in[22];
    const float* g_bn2 = (const float*)d_in[23]; const float* be_bn2 = (const float*)d_in[24];
    const float* m_bn2 = (const float*)d_in[25]; const float* v_bn2  = (const float*)d_in[26];
    const float* w_e   = (const float*)d_in[27]; const float* b_e  = (const float*)d_in[28];
    const float* w_a2  = (const float*)d_in[29]; const float* b_a2 = (const float*)d_in[30];
    float* out = (float*)d_out;

    char* ws = (char*)d_ws;
    size_t off = 0;
    auto alloc = [&](size_t bytes) { size_t o = off; off += (bytes + 255) & ~(size_t)255; return o; };
    const size_t NTOT = (size_t)B * NPTS;
    float* qf     = (float*)(ws + alloc(NTOT * DIM * 4));
    float* ff     = (float*)(ws + alloc(NTOT * DIM * 4));
    bf16*  keyT   = (bf16*) (ws + alloc(NTOT * DIM * 2));
    bf16*  queryT = (bf16*) (ws + alloc(NTOT * DIM * 2));
    float* valueT = (float*)(ws + alloc(NTOT * DIM * 4));
    int*   idx    = (int*)  (ws + alloc(NTOT * KNN * 4));
    bf16*  w_a1b  = (bf16*) (ws + alloc(HID * DIM * 2));
    bf16*  w_a2b  = (bf16*) (ws + alloc(HID * DIM * 2));
    bf16*  w_eb   = (bf16*) (ws + alloc(HID * DIM * 2));
    bf16*  w_p2b  = (bf16*) (ws + alloc(DIM * PH * 2));
    float* scale1 = (float*)(ws + alloc(PH * 4));
    float* bias1e = (float*)(ws + alloc(PH * 4));
    float* scale2 = (float*)(ws + alloc(HID * 4));
    float* bias2e = (float*)(ws + alloc(HID * 4));

    prep_kernel<<<256, 256, 0, stream>>>(
        w_a1, w_a2, w_e, w_p2, b_p1, g_bn1, be_bn1, m_bn1, v_bn1,
        b_a1, g_bn2, be_bn2, m_bn2, v_bn2,
        w_a1b, w_a2b, w_eb, w_p2b, scale1, bias1e, scale2, bias2e);

    knn_kernel<<<B * (NPTS / 8), 256, 0, stream>>>(pos, idx);

    const int gemm_blocks = (int)(NTOT / 32);
    // qf = w_sq @ query_points + b_sq   (K=512)
    gemm128_kernel<<<gemm_blocks, 256, 0, stream>>>(w_sq, qpts, b_sq, 512, qf, nullptr, nullptr);
    // f  = w_so @ obj_feature + b_so    (K=512)
    gemm128_kernel<<<gemm_blocks, 256, 0, stream>>>(w_so, obj,  b_so, 512, ff, nullptr, nullptr);
    // key = w_k @ f (transposed bf16), query = w_q @ qf (transposed bf16),
    // value = w_v @ f (transposed f32)
    gemm128_kernel<<<gemm_blocks, 256, 0, stream>>>(w_k, ff, b_k, 128, nullptr, keyT, nullptr);
    gemm128_kernel<<<gemm_blocks, 256, 0, stream>>>(w_q, qf, b_q, 128, nullptr, queryT, nullptr);
    gemm128_kernel<<<gemm_blocks, 256, 0, stream>>>(w_v, ff, b_v, 128, nullptr, nullptr, valueT);

    fused_attn_kernel<<<(int)(NTOT / 32), 256, 0, stream>>>(
        pos, idx, keyT, queryT, valueT,
        w_p2b, b_p2, w_p1, scale1, bias1e,
        w_a1b, scale2, bias2e, w_a2b, b_a2, w_eb, b_e, out);
}
